// TypeConditionedTemporalModel_21646635172623
// MI455X (gfx1250) — matrix operand order repaired and checked by reading, not compiled, not run on hardware
//
#include <hip/hip_runtime.h>
#include <hip/hip_bf16.h>

// ---------------------------------------------------------------------------
// Type-conditioned single-step LSTM for MI455X (gfx1250).
//
// out = LSTM_t(x,h,c) where t = LAST type with agent_types[b,t] > 0
// (zeros if none) -> one gate GEMV per row, at 16-row-tile granularity.
//
// Compute: V_WMMA_F32_16X16X32_BF16 with bf16 hi/lo split of activations and
// weights (3 products kept) -> ~fp32 accuracy at bf16 matrix-core rate.
// K = I+H = 128 fused (x|h vs [W_ih;W_hh]^T), 4 K-chunks of 32.
// Per-type weights are converted into WMMA B-fragment layout in LDS (128 KB
// of the WGP's 320 KB) and reused across the WG's 1024 rows.
// Nonlinearities use the CDNA5 hardware v_tanh_f32 (sigmoid via tanh
// identity) and are EXEC-predicated to only the rows actually selected.
// ---------------------------------------------------------------------------

typedef __attribute__((ext_vector_type(16))) __bf16 v16bf;
typedef __attribute__((ext_vector_type(8)))  float  v8f;

#define B_TOTAL      262144
#define HID          64
#define NTYPES       8
#define ROWS_PER_WG  1024
#define NUM_WG       (B_TOTAL / ROWS_PER_WG)      // 256
#define TILES_PER_WG (ROWS_PER_WG / 16)           // 64
#define BLOCK        256                           // 8 wave32s

struct SMem {
    // [split hi/lo][k-chunk][n-tile][lane][elem]  = 131072 B
    __bf16   wfrag[2][4][16][32][16];
    float    bias[256];                // b_ih + b_hh for current type
    short    sel[ROWS_PER_WG];
    unsigned tmask[TILES_PER_WG];
    unsigned wgmask;
};

// CDNA5 hardware transcendental tanh (TRANS op, co-executes with WMMA).
__device__ __forceinline__ float tanh_hw(float v) {
    float r;
    asm volatile("v_tanh_f32 %0, %1" : "=v"(r) : "v"(v));
    return r;
}
__device__ __forceinline__ float sigmoid_hw(float v) {
    return fmaf(0.5f, tanh_hw(0.5f * v), 0.5f);   // 1/(1+e^-v)
}
__device__ __forceinline__ void split_bf16(float v, __bf16& hi, __bf16& lo) {
    hi = (__bf16)v;
    lo = (__bf16)(v - (float)hi);
}

__global__ __launch_bounds__(BLOCK)
void lstm_type_cond_kernel(const float* __restrict__ x,
                           const float* __restrict__ h0,
                           const float* __restrict__ c0,
                           const int*   __restrict__ at,
                           const float* __restrict__ Wih,
                           const float* __restrict__ Whh,
                           const float* __restrict__ bih,
                           const float* __restrict__ bhh,
                           float*       __restrict__ out)
{
    __shared__ SMem sm;
    const int tid    = threadIdx.x;
    const int wave   = tid >> 5;
    const int lane   = tid & 31;
    const int m      = lane & 15;     // row within tile (A rows / C columns)
    const int laneHi = lane >> 4;
    const int wgBase = blockIdx.x * ROWS_PER_WG;

    float* out_y = out;
    float* out_h = out + (size_t)B_TOTAL * HID;
    float* out_c = out + (size_t)B_TOTAL * HID * 2;

    if (tid < TILES_PER_WG) sm.tmask[tid] = 0u;
    if (tid == 0)           sm.wgmask    = 0u;
    __syncthreads();

    // ---- prepass: sel = last positive type per row; zero rows with none ----
    #pragma unroll
    for (int kb = 0; kb < ROWS_PER_WG / BLOCK; ++kb) {
        const int rl = kb * BLOCK + tid;
        const int r  = wgBase + rl;
        int s = -1;
        #pragma unroll
        for (int t = 0; t < NTYPES; ++t)
            if (at[(size_t)r * NTYPES + t] > 0) s = t;
        sm.sel[rl] = (short)s;
        if (s >= 0) {
            atomicOr(&sm.tmask[rl >> 4], 1u << s);
            atomicOr(&sm.wgmask, 1u << s);
        } else {
            const float4 z = make_float4(0.f, 0.f, 0.f, 0.f);
            float4* py = (float4*)(out_y + (size_t)r * HID);
            float4* ph = (float4*)(out_h + (size_t)r * HID);
            float4* pc = (float4*)(out_c + (size_t)r * HID);
            #pragma unroll
            for (int q = 0; q < HID / 4; ++q) { py[q] = z; ph[q] = z; pc[q] = z; }
        }
    }
    __syncthreads();
    const unsigned wgmask = sm.wgmask;   // uniform across the WG

    for (int t = 0; t < NTYPES; ++t) {
        if (!((wgmask >> t) & 1u)) continue;       // uniform skip

        __syncthreads();  // previous type's LDS reads complete
        // ---- stage type-t weights into WMMA B-fragment layout (hi+lo) ----
        {
            const float* Wi = Wih + (size_t)t * 256 * 64;
            const float* Wh = Whh + (size_t)t * 256 * 64;
            for (int idx = tid; idx < 4 * 16 * 32 * 16; idx += BLOCK) {
                const int e  = idx & 15;
                const int ln = (idx >> 4) & 31;
                const int nt = (idx >> 9) & 15;
                const int kc = idx >> 13;
                const int nn = nt * 16 + (ln & 15);        // gate-output col 0..255
                const int kk = ((ln >> 4) << 4) + e;       // lanes 16-31 hold K 16..31
                const int k  = kc * 32 + kk;               // combined K 0..127
                const float w = (k < 64) ? Wi[nn * 64 + k] : Wh[nn * 64 + (k - 64)];
                __bf16 hi, lo; split_bf16(w, hi, lo);
                sm.wfrag[0][kc][nt][ln][e] = hi;
                sm.wfrag[1][kc][nt][ln][e] = lo;
            }
            sm.bias[tid] = bih[t * 256 + tid] + bhh[t * 256 + tid];
        }
        __syncthreads();

        for (int tile = wave; tile < TILES_PER_WG; tile += BLOCK / 32) {
            if (!((sm.tmask[tile] >> t) & 1u)) continue;   // uniform per wave
            const int rowBase = wgBase + tile * 16;

            // sel for this lane's 8 candidate rows: one packed 16B LDS load
            unsigned selw[4];
            {
                const uint4 sp = *(const uint4*)&sm.sel[tile * 16 + laneHi * 8];
                selw[0] = sp.x; selw[1] = sp.y; selw[2] = sp.z; selw[3] = sp.w;
            }

            // ---- A fragments (x|h hi/lo), contiguous K order (half h: K 16h..16h+15), same as B ----
            v16bf ahi[4], alo[4];
            {
                const float* xr = x  + (size_t)(rowBase + m) * HID;
                const float* hr = h0 + (size_t)(rowBase + m) * HID;
                #pragma unroll
                for (int kc = 0; kc < 4; ++kc) {
                    const float* src = (kc < 2) ? xr : hr;
                    const int kb = (kc & 1) * 32 + laneHi * 16;
                    #pragma unroll
                    for (int run = 0; run < 2; ++run) {    // K runs {0..7},{8..15} (+16 hi)
                        const float4 f0 = *(const float4*)(src + kb + run * 8);
                        const float4 f1 = *(const float4*)(src + kb + run * 8 + 4);
                        const float vals[8] = { f0.x, f0.y, f0.z, f0.w,
                                                f1.x, f1.y, f1.z, f1.w };
                        #pragma unroll
                        for (int q = 0; q < 8; ++q) {
                            __bf16 hi, lo; split_bf16(vals[q], hi, lo);
                            ahi[kc][run * 8 + q] = hi;
                            alo[kc][run * 8 + q] = lo;
                        }
                    }
                }
            }

            #pragma unroll
            for (int j = 0; j < 4; ++j) {          // 16-column block of each gate
                v8f acc[4];
                #pragma unroll
                for (int g = 0; g < 4; ++g) acc[g] = (v8f)(0.0f);

                #pragma unroll
                for (int kc = 0; kc < 4; ++kc) {
                    #pragma unroll
                    for (int g = 0; g < 4; ++g) {  // i,f,g,o
                        const int nt = g * 4 + j;
                        const v16bf bhi = *(const v16bf*)&sm.wfrag[0][kc][nt][lane][0];
                        const v16bf blo = *(const v16bf*)&sm.wfrag[1][kc][nt][lane][0];
                        acc[g] = __builtin_amdgcn_wmma_f32_16x16x32_bf16(
                                     false, ahi[kc], false, bhi, (short)0, acc[g], false, false);
                        acc[g] = __builtin_amdgcn_wmma_f32_16x16x32_bf16(
                                     false, ahi[kc], false, blo, (short)0, acc[g], false, false);
                        acc[g] = __builtin_amdgcn_wmma_f32_16x16x32_bf16(
                                     false, alo[kc], false, bhi, (short)0, acc[g], false, false);
                    }
                }

                const int ncol = j * 16 + (lane & 15);
                float bs[4];
                #pragma unroll
                for (int g = 0; g < 4; ++g)
                    bs[g] = sm.bias[g * 64 + ncol];

                #pragma unroll
                for (int vg = 0; vg < 8; ++vg) {   // C layout: rows vg and vg+8
                    const int selv = (int)(short)((selw[vg >> 1] >> ((vg & 1) * 16)) & 0xFFFFu);
                    if (selv == t) {               // EXEC-predicated: skip dead rows
                        const int r  = rowBase + vg + laneHi * 8;
                        const float gi = sigmoid_hw(acc[0][vg] + bs[0]);
                        const float gf = sigmoid_hw(acc[1][vg] + bs[1]);
                        const float gg = tanh_hw   (acc[2][vg] + bs[2]);
                        const float go = sigmoid_hw(acc[3][vg] + bs[3]);
                        const float cp = c0[(size_t)r * HID + ncol];
                        const float ct = gf * cp + gi * gg;
                        const float ht = go * tanh_hw(ct);
                        out_y[(size_t)r * HID + ncol] = ht;
                        out_h[(size_t)r * HID + ncol] = ht;
                        out_c[(size_t)r * HID + ncol] = ct;
                    }
                }
            }
        }
    }
}

extern "C" void kernel_launch(void* const* d_in, const int* in_sizes, int n_in,
                              void* d_out, int out_size, void* d_ws, size_t ws_size,
                              hipStream_t stream) {
    (void)in_sizes; (void)n_in; (void)out_size; (void)d_ws; (void)ws_size;
    const float* x   = (const float*)d_in[0];
    const float* h0  = (const float*)d_in[1];
    const float* c0  = (const float*)d_in[2];
    const int*   at  = (const int*)  d_in[3];
    const float* Wih = (const float*)d_in[4];
    const float* Whh = (const float*)d_in[5];
    const float* bih = (const float*)d_in[6];
    const float* bhh = (const float*)d_in[7];
    lstm_type_cond_kernel<<<NUM_WG, BLOCK, 0, stream>>>(
        x, h0, c0, at, Wih, Whh, bih, bhh, (float*)d_out);
}
